// Detect_21466246545878
// MI455X (gfx1250) — matrix operand order repaired and checked by reading, not compiled, not run on hardware
//
#include <hip/hip_runtime.h>
#include <hip/hip_bf16.h>
#include <math.h>

// ---------------------------------------------------------------------------
// CDNA5 (gfx1250) YOLO-seg head.
// Conv GEMMs: v_wmma_f32_16x16x32_f16, 64Mx64N block tile, 4 WMMA / wave / K-step.
// Activation tiles (1x1 convs) streamed with global_load_async_to_lds_b128.
// ---------------------------------------------------------------------------

typedef __attribute__((ext_vector_type(16))) _Float16 v16h;
typedef __attribute__((ext_vector_type(8)))  _Float16 v8h;
typedef __attribute__((ext_vector_type(8)))  float    v8f;

#define NC_      80
#define NO_      85
#define NPRED    25200
#define MAXDET   100
#define TOPK_    1024
#define CONF_T   0.15f
#define IOU_T    0.45f
#define MAX_WH_  4096.0f

#define SA_STRIDE 68   // f32 per K-row: 64 M + 4 pad (272B, 16B aligned, bank-skewed)
#define SB_STRIDE 40   // f16 per N-row: 32 K + 8 pad (80B, 16B aligned, bank-skewed)

__constant__ float c_anch[3][3][2] = {
    {{10.f, 13.f}, {16.f, 30.f}, {33.f, 23.f}},
    {{30.f, 61.f}, {62.f, 45.f}, {59.f, 119.f}},
    {{116.f, 90.f}, {156.f, 198.f}, {373.f, 326.f}}};

__device__ inline float sigm(float x) { return 1.0f / (1.0f + __expf(-x)); }

// ---- WMMA fragment builders (16-bit, contiguous k order, wave32) ------------
// A 16x32: lane L -> row m = mcol + (L&15); halfs h -> K=koff+h,
// koff = 16*(L>>4).  sA is f32 [K][SA_STRIDE].
__device__ inline v16h frag_a_f32lds(const float* sA, int mcol, int lane) {
  int m = mcol + (lane & 15);
  int koff = (lane >> 4) << 4;
  v16h a;
#pragma unroll
  for (int h = 0; h < 16; ++h) {
    int k = koff + h;
    a[h] = (_Float16)sA[k * SA_STRIDE + m];
  }
  return a;
}
// B 32x16: lane L -> col n; halfs h -> K = h + 16*(L>>4).
// sB is f16 [N][SB_STRIDE] so a fragment is two aligned 16B LDS loads.
__device__ inline v16h frag_b_f16lds(const _Float16* sB, int lane, int ntile) {
  int n = ntile * 16 + (lane & 15);
  int kb = (lane >> 4) << 4;
  const v8h* p = (const v8h*)(sB + n * SB_STRIDE + kb);
  v8h lo = p[0];
  v8h hi = p[1];
  return __builtin_shufflevector(lo, hi, 0, 1, 2, 3, 4, 5, 6, 7,
                                 8, 9, 10, 11, 12, 13, 14, 15);
}

__device__ inline void async_lds_b128(const float* gp, const void* lp) {
  unsigned lds = (unsigned)(unsigned long long)lp;  // LDS offset = low 32 bits
  asm volatile("global_load_async_to_lds_b128 %0, %1, off"
               :: "v"(lds), "v"(gp)
               : "memory");
}
__device__ inline void wait_async0() {
  asm volatile("s_wait_asynccnt 0x0" ::: "memory");
}

// ---------------------------------------------------------------------------
// 1x1 conv GEMM: out[b,n,p] = sum_k x[b,k,p]*w[n,k] + bias[n]
// Block 128 thr (4 waves); tile 64M x 64N; A via async DMA to LDS.
// ---------------------------------------------------------------------------
__global__ void conv1x1_wmma_kernel(const float* __restrict__ x,
                                    const float* __restrict__ w,
                                    const float* __restrict__ bias,
                                    float* __restrict__ out,
                                    int C, int N, int HW) {
  __shared__ float    sA[32 * SA_STRIDE];
  __shared__ _Float16 sB[64 * SB_STRIDE];
  int tid = threadIdx.x, lane = tid & 31, wave = tid >> 5;
  int m0 = blockIdx.x * 64, n0 = blockIdx.y * 64, b = blockIdx.z;
  const float* xb = x + (size_t)b * C * HW;
  v8f acc[4] = {};
  for (int k0 = 0; k0 < C; k0 += 32) {
    // A tile: 32K x 64M f32 via async-to-LDS (4 x b128 per thread)
#pragma unroll
    for (int j = 0; j < 4; ++j) {
      int idx = tid + j * 128;        // 512 float4 transfers
      int k = idx >> 4;               // 0..31
      int mq = (idx & 15) << 2;       // 0..60
      int mg = m0 + mq;
      if (mg > HW - 4) mg = HW - 4;   // clamp tail (rows discarded at store)
      async_lds_b128(xb + (size_t)(k0 + k) * HW + mg, &sA[k * SA_STRIDE + mq]);
    }
    // B tile: 64N x 32K f16, one b128 LDS store per 8 halfs
#pragma unroll
    for (int j = 0; j < 2; ++j) {
      int idx = tid + j * 128;        // 256 jobs
      int n = idx >> 2;
      int k8 = (idx & 3) << 3;
      int nn = n0 + n;
      v8h hv;
#pragma unroll
      for (int q = 0; q < 8; ++q)
        hv[q] = (nn < N) ? (_Float16)w[(size_t)nn * C + k0 + k8 + q]
                         : (_Float16)0.0f;
      *(v8h*)(&sB[n * SB_STRIDE + k8]) = hv;
    }
    if (k0 + 32 < C)
      __builtin_prefetch(&w[(size_t)(n0 + (tid & 63)) * C + k0 + 32], 0, 1);
    wait_async0();
    __syncthreads();
    v16h a = frag_a_f32lds(sA, wave * 16, lane);
#pragma unroll
    for (int t = 0; t < 4; ++t) {
      v16h bf = frag_b_f16lds(sB, lane, t);
      acc[t] = __builtin_amdgcn_wmma_f32_16x16x32_f16(false, a, false, bf,
                                                      (short)0, acc[t],
                                                      false, false);
    }
    __syncthreads();
  }
  int mbase = m0 + wave * 16 + ((lane >> 4) << 3);
#pragma unroll
  for (int t = 0; t < 4; ++t) {
    int n = n0 + t * 16 + (lane & 15);
    if (n < N) {
      float bv = bias ? bias[n] : 0.0f;
      float* ob = out + ((size_t)b * N + n) * HW;
#pragma unroll
      for (int r = 0; r < 8; ++r) {
        int m = mbase + r;
        if (m < HW) ob[m] = acc[t][r] + bv;
      }
    }
  }
}

// ---------------------------------------------------------------------------
// 3x3 SAME conv-bn (im2col GEMM, gamma folded), optional SiLU.
// Same 64x64 tiling; A gathered synchronously (scattered im2col addresses).
// ---------------------------------------------------------------------------
__global__ void conv3x3_wmma_kernel(const float* __restrict__ x,
                                    const float* __restrict__ w,
                                    const float* __restrict__ g,
                                    const float* __restrict__ bias,
                                    float* __restrict__ out,
                                    int C, int N, int H, int W, int act) {
  __shared__ float    sA[32 * SA_STRIDE];
  __shared__ _Float16 sB[64 * SB_STRIDE];
  int tid = threadIdx.x, lane = tid & 31, wave = tid >> 5;
  int m0 = blockIdx.x * 64, n0 = blockIdx.y * 64, b = blockIdx.z;
  int K = C * 9, HW = H * W;
  const float* xb = x + (size_t)b * C * HW;
  v8f acc[4] = {};
  for (int k0 = 0; k0 < K; k0 += 32) {
    // A tile: im2col gather, f32 into [K][M] rows
#pragma unroll 4
    for (int j = 0; j < 16; ++j) {
      int idx = tid + j * 128;        // 2048 elements
      int m = idx & 63;
      int k = idx >> 6;
      int kk = k0 + k;
      int c = kk / 9, j9 = kk - c * 9;
      int ky = j9 / 3, kx = j9 - ky * 3;
      int p = m0 + m;
      float v = 0.0f;
      if (p < HW) {
        int py = p / W + ky - 1;
        int px = p - (p / W) * W + kx - 1;
        if (py >= 0 && py < H && px >= 0 && px < W)
          v = xb[((size_t)c * H + py) * W + px];
      }
      sA[k * SA_STRIDE + m] = v;
    }
    // B tile (gamma folded into weights)
#pragma unroll
    for (int j = 0; j < 2; ++j) {
      int idx = tid + j * 128;
      int n = idx >> 2;
      int k8 = (idx & 3) << 3;
      int nn = n0 + n;
      float gv = (nn < N && g) ? g[nn] : 1.0f;
      v8h hv;
#pragma unroll
      for (int q = 0; q < 8; ++q)
        hv[q] = (nn < N) ? (_Float16)(w[(size_t)nn * K + k0 + k8 + q] * gv)
                         : (_Float16)0.0f;
      *(v8h*)(&sB[n * SB_STRIDE + k8]) = hv;
    }
    if (k0 + 32 < K)
      __builtin_prefetch(&w[(size_t)(n0 + (tid & 63)) * K + k0 + 32], 0, 1);
    __syncthreads();
    v16h a = frag_a_f32lds(sA, wave * 16, lane);
#pragma unroll
    for (int t = 0; t < 4; ++t) {
      v16h bf = frag_b_f16lds(sB, lane, t);
      acc[t] = __builtin_amdgcn_wmma_f32_16x16x32_f16(false, a, false, bf,
                                                      (short)0, acc[t],
                                                      false, false);
    }
    __syncthreads();
  }
  int mbase = m0 + wave * 16 + ((lane >> 4) << 3);
#pragma unroll
  for (int t = 0; t < 4; ++t) {
    int n = n0 + t * 16 + (lane & 15);
    if (n < N) {
      float bv = bias ? bias[n] : 0.0f;
      float* ob = out + ((size_t)b * N + n) * HW;
#pragma unroll
      for (int r = 0; r < 8; ++r) {
        int m = mbase + r;
        if (m < HW) {
          float v = acc[t][r] + bv;
          if (act) v = v * (1.0f / (1.0f + __expf(-v)));  // SiLU
          ob[m] = v;
        }
      }
    }
  }
}

// ---------------------------------------------------------------------------
// Decode: fraw [B,255,ny,nx] -> preds [B,25200,85]
// ---------------------------------------------------------------------------
__global__ void decode_kernel(const float* __restrict__ fraw,
                              float* __restrict__ preds,
                              int ny, int nx, float stride, int scale, int base) {
  int total = 2 * 3 * ny * nx;
  int idx = blockIdx.x * blockDim.x + threadIdx.x;
  if (idx >= total) return;
  int hw = ny * nx;
  int p = idx % hw;
  int t = idx / hw;
  int a = t % 3, b = t / 3;
  int gy = p / nx, gx = p - gy * nx;
  size_t cs = (size_t)hw;
  const float* fb = fraw + (((size_t)b * 255 + a * NO_) * ny + gy) * nx + gx;
  float* o = preds + ((size_t)b * NPRED + base + (size_t)a * hw + p) * NO_;
  float s0 = sigm(fb[0 * cs]), s1 = sigm(fb[1 * cs]);
  float s2 = sigm(fb[2 * cs]), s3 = sigm(fb[3 * cs]);
  o[0] = (2.0f * s0 + (float)gx - 0.5f) * stride;
  o[1] = (2.0f * s1 + (float)gy - 0.5f) * stride;
  float tw = 2.0f * s2, th = 2.0f * s3;
  o[2] = tw * tw * c_anch[scale][a][0];
  o[3] = th * th * c_anch[scale][a][1];
#pragma unroll 4
  for (int k = 4; k < NO_; ++k) o[k] = sigm(fb[(size_t)k * cs]);
}

// ---------------------------------------------------------------------------
// Align-corners bilinear resize + skip add
// ---------------------------------------------------------------------------
__global__ void resize_add_kernel(const float* __restrict__ in,
                                  const float* __restrict__ skip,
                                  float* __restrict__ out,
                                  int B, int C, int H, int W, int OH, int OW) {
  int total = B * C * OH * OW;
  int idx = blockIdx.x * blockDim.x + threadIdx.x;
  if (idx >= total) return;
  int ox = idx % OW;
  int t = idx / OW;
  int oy = t % OH;
  t /= OH;
  int c = t % C, b = t / C;
  float sy = (float)(H - 1) / (float)(OH - 1);
  float sx = (float)(W - 1) / (float)(OW - 1);
  float yf = oy * sy, xf = ox * sx;
  int y0 = (int)floorf(yf), x0 = (int)floorf(xf);
  int y1 = min(y0 + 1, H - 1), x1 = min(x0 + 1, W - 1);
  float fy = yf - y0, fx = xf - x0;
  const float* ib = in + ((size_t)b * C + c) * H * W;
  float r0 = ib[y0 * W + x0] * (1.0f - fy) + ib[y1 * W + x0] * fy;
  float r1 = ib[y0 * W + x1] * (1.0f - fy) + ib[y1 * W + x1] * fy;
  out[idx] = r0 * (1.0f - fx) + r1 * fx + skip[idx];
}

// ---------------------------------------------------------------------------
// NMS: one 1024-thread block per image.
// ---------------------------------------------------------------------------
__device__ inline float iou_box(const float* a, const float* b) {
  float lx = fmaxf(a[0], b[0]), ly = fmaxf(a[1], b[1]);
  float rx = fminf(a[2], b[2]), ry = fminf(a[3], b[3]);
  float iw = fmaxf(rx - lx, 0.0f), ih = fmaxf(ry - ly, 0.0f);
  float inter = iw * ih;
  float aa = (a[2] - a[0]) * (a[3] - a[1]);
  float ab = (b[2] - b[0]) * (b[3] - b[1]);
  return inter / (aa + ab - inter + 1e-7f);
}

__global__ void nms_kernel(const float* __restrict__ preds,
                           float* __restrict__ out_boxes,
                           float* __restrict__ out_scores,
                           float* __restrict__ out_labels,
                           float* __restrict__ out_valid,
                           float* __restrict__ rois) {
  __shared__ float s_score[TOPK_];
  __shared__ float s_box[TOPK_][4];
  __shared__ int s_cls[TOPK_];
  __shared__ unsigned char s_supp[TOPK_];
  __shared__ unsigned char s_keep[TOPK_];
  __shared__ int s_cnt;

  int b = blockIdx.x, tid = threadIdx.x;
  if (tid == 0) s_cnt = 0;
  s_score[tid] = -1.0f;
  s_cls[tid] = 0;
  s_supp[tid] = 0;
  s_keep[tid] = 0;
  s_box[tid][0] = s_box[tid][1] = s_box[tid][2] = s_box[tid][3] = 0.0f;
  __syncthreads();

  const float* P = preds + (size_t)b * NPRED * NO_;
  for (int i = tid; i < NPRED; i += TOPK_) {
    const float* p = P + (size_t)i * NO_;
    float obj = p[4];
    float mx = p[5];
    int mi = 0;
    for (int c = 1; c < NC_; ++c) {
      float v = p[5 + c];
      if (v > mx) { mx = v; mi = c; }
    }
    float sc = obj * mx;
    if (sc > CONF_T) {
      int pos = atomicAdd(&s_cnt, 1);
      if (pos < TOPK_) {
        float cx = p[0], cy = p[1], w = p[2], h = p[3];
        float off = (float)mi * MAX_WH_;
        s_box[pos][0] = cx - 0.5f * w + off;
        s_box[pos][1] = cy - 0.5f * h + off;
        s_box[pos][2] = cx + 0.5f * w + off;
        s_box[pos][3] = cy + 0.5f * h + off;
        s_score[pos] = sc;
        s_cls[pos] = mi;
      }
    }
  }
  __syncthreads();

  // Bitonic sort, descending by score
  for (int k = 2; k <= TOPK_; k <<= 1) {
    for (int j = k >> 1; j > 0; j >>= 1) {
      int i = tid, ixj = i ^ j;
      if (ixj > i) {
        bool dir = ((i & k) == 0);
        bool sw = dir ? (s_score[i] < s_score[ixj]) : (s_score[i] > s_score[ixj]);
        if (sw) {
          float ts = s_score[i]; s_score[i] = s_score[ixj]; s_score[ixj] = ts;
          int tc = s_cls[i]; s_cls[i] = s_cls[ixj]; s_cls[ixj] = tc;
#pragma unroll
          for (int q = 0; q < 4; ++q) {
            float tb = s_box[i][q]; s_box[i][q] = s_box[ixj][q]; s_box[ixj][q] = tb;
          }
        }
      }
      __syncthreads();
    }
  }

  // Greedy suppression; thread tid owns column j=tid.
  for (int i = 0; i < TOPK_; ++i) {
    __syncthreads();
    bool ki = (s_score[i] > CONF_T) && (s_supp[i] == 0);
    if (tid == 0) s_keep[i] = ki ? 1 : 0;
    __syncthreads();
    if (ki) {
      if (iou_box(&s_box[i][0], &s_box[tid][0]) > IOU_T) s_supp[tid] = 1;
    }
  }
  __syncthreads();

  if (tid == 0) {
    int outc = 0;
    for (int i = 0; i < TOPK_ && outc < MAXDET; ++i) {
      if (s_keep[i]) {
        float off = (float)s_cls[i] * MAX_WH_;
        int o = b * MAXDET + outc;
        float bx[4];
#pragma unroll
        for (int q = 0; q < 4; ++q) bx[q] = s_box[i][q] - off;
#pragma unroll
        for (int q = 0; q < 4; ++q) {
          out_boxes[o * 4 + q] = bx[q];
          rois[o * 4 + q] = bx[q] * 0.125f;
        }
        out_scores[o] = s_score[i];
        out_labels[o] = (float)(s_cls[i] + 1);
        out_valid[o] = 1.0f;
        ++outc;
      }
    }
    for (; outc < MAXDET; ++outc) {
      int o = b * MAXDET + outc;
#pragma unroll
      for (int q = 0; q < 4; ++q) { out_boxes[o * 4 + q] = 0.0f; rois[o * 4 + q] = 0.0f; }
      out_scores[o] = 0.0f;
      out_labels[o] = 0.0f;
      out_valid[o] = 0.0f;
    }
  }
}

// ---------------------------------------------------------------------------
// ROI-align 28x28 + fused SiLU -> crops [200,128,784]
// ---------------------------------------------------------------------------
__global__ void roi_align_silu_kernel(const float* __restrict__ f,
                                      const float* __restrict__ rois,
                                      float* __restrict__ crops) {
  int c = blockIdx.x;
  int d = blockIdx.y;
  int b = d / MAXDET;
  const float* img = f + ((size_t)(b * 128 + c)) * 80 * 80;
  float x1 = rois[d * 4 + 0], y1 = rois[d * 4 + 1];
  float x2 = rois[d * 4 + 2], y2 = rois[d * 4 + 3];
  float rw = fmaxf(x2 - x1, 1.0f), rh = fmaxf(y2 - y1, 1.0f);
  for (int p = threadIdx.x; p < 784; p += blockDim.x) {
    int oy = p / 28, ox = p - oy * 28;
    float cy = y1 + ((float)oy + 0.5f) * (1.0f / 28.0f) * rh;
    float cx = x1 + ((float)ox + 0.5f) * (1.0f / 28.0f) * rw;
    cy = fminf(fmaxf(cy, 0.0f), 79.0f);
    cx = fminf(fmaxf(cx, 0.0f), 79.0f);
    int y0 = (int)floorf(cy), x0 = (int)floorf(cx);
    int yy1 = min(y0 + 1, 79), xx1 = min(x0 + 1, 79);
    float ly = cy - y0, lx = cx - x0;
    float v = img[y0 * 80 + x0] * (1.0f - ly) * (1.0f - lx) +
              img[y0 * 80 + xx1] * (1.0f - ly) * lx +
              img[yy1 * 80 + x0] * ly * (1.0f - lx) +
              img[yy1 * 80 + xx1] * ly * lx;
    crops[((size_t)d * 128 + c) * 784 + p] = v * (1.0f / (1.0f + __expf(-v)));
  }
}

// ---------------------------------------------------------------------------
// Mask head: selected-class channel only (128-dot per pixel).
// ---------------------------------------------------------------------------
__global__ void mask_kernel(const float* __restrict__ h,
                            const float* __restrict__ w_ml,
                            const float* __restrict__ b_ml,
                            const float* __restrict__ labels,
                            const float* __restrict__ valid,
                            float* __restrict__ masks) {
  __shared__ float sw[128];
  int d = blockIdx.x;
  int lab = (int)labels[d] - 1;
  lab = max(0, min(lab, NC_ - 1));
  if (threadIdx.x < 128) sw[threadIdx.x] = w_ml[(size_t)lab * 128 + threadIdx.x];
  __syncthreads();
  float bv = b_ml[lab];
  float vld = valid[d];
  const float* hb = h + (size_t)d * 128 * 784;
  for (int p = threadIdx.x; p < 784; p += blockDim.x) {
    float acc = bv;
#pragma unroll 8
    for (int c = 0; c < 128; ++c) acc += hb[(size_t)c * 784 + p] * sw[c];
    masks[(size_t)d * 784 + p] = (1.0f / (1.0f + __expf(-acc))) * vld;
  }
}

// ---------------------------------------------------------------------------
extern "C" void kernel_launch(void* const* d_in, const int* in_sizes, int n_in,
                              void* d_out, int out_size, void* d_ws, size_t ws_size,
                              hipStream_t stream) {
  (void)in_sizes; (void)n_in; (void)out_size; (void)ws_size;
  const float* x0 = (const float*)d_in[0];
  const float* x1 = (const float*)d_in[1];
  const float* x2 = (const float*)d_in[2];
  const float* w_det0 = (const float*)d_in[3];
  const float* b_det0 = (const float*)d_in[4];
  const float* w_det1 = (const float*)d_in[5];
  const float* b_det1 = (const float*)d_in[6];
  const float* w_det2 = (const float*)d_in[7];
  const float* b_det2 = (const float*)d_in[8];
  const float* w_seg0 = (const float*)d_in[9];
  const float* g_seg0 = (const float*)d_in[10];
  const float* b_seg0 = (const float*)d_in[11];
  const float* w_seg1 = (const float*)d_in[12];
  const float* g_seg1 = (const float*)d_in[13];
  const float* b_seg1 = (const float*)d_in[14];
  const float* w_seg2 = (const float*)d_in[15];
  const float* g_seg2 = (const float*)d_in[16];
  const float* b_seg2 = (const float*)d_in[17];
  const float* w_sc = (const float*)d_in[18];
  const float* g_sc = (const float*)d_in[19];
  const float* b_sc = (const float*)d_in[20];
  const float* w_ml = (const float*)d_in[21];
  const float* b_ml = (const float*)d_in[22];

  float* out = (float*)d_out;
  float* out_boxes = out;
  float* out_scores = out + 800;
  float* out_labels = out + 1000;
  float* out_valid = out + 1200;
  float* out_masks = out + 1400;

  float* W = (float*)d_ws;
  size_t o = 0;
  float* fraw0 = W + o; o += (size_t)2 * 255 * 6400;
  float* fraw1 = W + o; o += (size_t)2 * 255 * 1600;
  float* fraw2 = W + o; o += (size_t)2 * 255 * 400;
  float* preds = W + o; o += (size_t)2 * NPRED * NO_;
  float* t0    = W + o; o += (size_t)2 * 512 * 400;
  float* r1    = W + o; o += (size_t)2 * 512 * 1600;
  float* t1    = W + o; o += (size_t)2 * 256 * 1600;
  float* r2    = W + o; o += (size_t)2 * 256 * 6400;
  float* fseg  = W + o; o += (size_t)2 * 128 * 6400;
  float* rois  = W + o; o += (size_t)200 * 4;
  float* crops = W + o; o += (size_t)200 * 128 * 784;
  float* hsc   = W + o; o += (size_t)200 * 128 * 784;

  dim3 blk(128);

  // Detect heads (1x1 GEMMs, async A staging): grid = (ceil(HW/64), ceil(N/64), B)
  conv1x1_wmma_kernel<<<dim3(100, 4, 2), blk, 0, stream>>>(x0, w_det0, b_det0, fraw0, 256, 255, 6400);
  conv1x1_wmma_kernel<<<dim3(25, 4, 2), blk, 0, stream>>>(x1, w_det1, b_det1, fraw1, 512, 255, 1600);
  conv1x1_wmma_kernel<<<dim3(7, 4, 2), blk, 0, stream>>>(x2, w_det2, b_det2, fraw2, 1024, 255, 400);

  decode_kernel<<<(2 * 3 * 6400 + 255) / 256, 256, 0, stream>>>(fraw0, preds, 80, 80, 8.0f, 0, 0);
  decode_kernel<<<(2 * 3 * 1600 + 255) / 256, 256, 0, stream>>>(fraw1, preds, 40, 40, 16.0f, 1, 19200);
  decode_kernel<<<(2 * 3 * 400 + 255) / 256, 256, 0, stream>>>(fraw2, preds, 20, 20, 32.0f, 2, 24000);

  // Seg FPN
  conv3x3_wmma_kernel<<<dim3(7, 8, 2), blk, 0, stream>>>(x2, w_seg0, g_seg0, b_seg0, t0, 1024, 512, 20, 20, 0);
  resize_add_kernel<<<(2 * 512 * 1600 + 255) / 256, 256, 0, stream>>>(t0, x1, r1, 2, 512, 20, 20, 40, 40);
  conv3x3_wmma_kernel<<<dim3(25, 4, 2), blk, 0, stream>>>(r1, w_seg1, g_seg1, b_seg1, t1, 512, 256, 40, 40, 0);
  resize_add_kernel<<<(2 * 256 * 6400 + 255) / 256, 256, 0, stream>>>(t1, x0, r2, 2, 256, 40, 40, 80, 80);
  conv3x3_wmma_kernel<<<dim3(100, 2, 2), blk, 0, stream>>>(r2, w_seg2, g_seg2, b_seg2, fseg, 256, 128, 80, 80, 0);

  // NMS
  nms_kernel<<<dim3(2), dim3(TOPK_), 0, stream>>>(preds, out_boxes, out_scores, out_labels, out_valid, rois);

  // ROI align + SiLU
  roi_align_silu_kernel<<<dim3(128, 200), dim3(256), 0, stream>>>(fseg, rois, crops);

  // Crop conv (FLOP-dominant GEMM): grid = (ceil(784/64)=13, 2, 200)
  conv3x3_wmma_kernel<<<dim3(13, 2, 200), blk, 0, stream>>>(crops, w_sc, g_sc, b_sc, hsc, 128, 128, 28, 28, 1);

  // Mask head
  mask_kernel<<<dim3(200), dim3(256), 0, stream>>>(hsc, w_ml, b_ml, out_labels, out_valid, out_masks);
}
